// GPTLanguageModel_55027120996948
// MI455X (gfx1250) — compile-verified, matrix operand order checked by reading, not run on hardware
//
#include <hip/hip_runtime.h>
#include <hip/hip_bf16.h>

// ---------------------------------------------------------------------------
// GPT forward for MI455X (gfx1250, wave32, WMMA + Tensor Data Mover).
//  - all GEMMs: v_wmma_f32_16x16x32_f16, f32 accumulate
//  - weights pre-converted fp32 -> f16 and transposed to [N,K] once per launch
//  - GEMM mainloop: TDM tensor_load_to_lds double-buffered (BK=64),
//    gated with s_wait_tensorcnt + workgroup barriers
//  - V projection writes V^T so attention's P*V fragments are pair-contiguous
// ---------------------------------------------------------------------------

typedef __attribute__((ext_vector_type(16))) _Float16     v16h;
typedef __attribute__((ext_vector_type(8)))  float        v8f;
typedef __attribute__((ext_vector_type(8)))  int          v8i;
typedef __attribute__((ext_vector_type(4)))  int          v4i;
typedef __attribute__((ext_vector_type(4)))  unsigned int v4u;

union Frag16 { v16h h; v8i i; };

__device__ __forceinline__ v8f vzero8() {
  v8f z; for (int i = 0; i < 8; ++i) z[i] = 0.f; return z;
}

// 16-bit A/B fragment layout (ISA 7.12.2): register r holds the K-pair at:
__device__ __forceinline__ int kpair(int r, int grp) {
  return ((r & 4) << 2) + 8 * grp + ((r & 3) << 1);   // {0|16} + 8*grp + 2*(r%4)
}

__device__ __forceinline__ v8f wmma_f16(const Frag16& a, const Frag16& b, v8f c) {
  return __builtin_amdgcn_wmma_f32_16x16x32_f16(
      false, a.h, false, b.h, (short)0, c, false, false);
}

#define E_DIM   1024
#define T_DIM   1024
#define HS_DIM  64
#define DFF_DIM 4096
#define ROWS_DIM 2048      // B*T

// ---------------------------------------------------------------------------
// Tensor Data Mover support (toolchain-dependent builtin arity)
// ---------------------------------------------------------------------------
#if defined(__has_builtin)
#if __has_builtin(__builtin_amdgcn_tensor_load_to_lds)
#define HAVE_TDM 1
#endif
#endif
#ifndef HAVE_TDM
#define HAVE_TDM 0
#endif

#if HAVE_TDM
// 2D tile load: rows x tile_k f16 elements, row stride = stride_elems.
// D# built per CDNA5 ISA (group0 + group1, groups 2/3 zero).
__device__ __forceinline__ void tdm_load_tile(const _Float16* gsrc, unsigned int lds_addr,
                                              int tile_k, int rows,
                                              int stride_elems, int td0, int td1) {
  unsigned long long ga = (unsigned long long)(uintptr_t)gsrc;
  v4u g0;
  g0[0] = 1u;                                           // count=1, user desc
  g0[1] = lds_addr;                                     // lds_addr [63:32]
  g0[2] = (unsigned int)ga;                             // global_addr[31:0]
  g0[3] = (unsigned int)((ga >> 32) & 0x1FFFFFFull)     // global_addr[56:32]
          | (2u << 30);                                 // type=2 ("image")
  v8i g1;
  g1[0] = (int)(1u << 16);                              // data_size=1 (2 bytes)
  g1[1] = (int)(((unsigned)td0 & 0xFFFFu) << 16);       // tensor_dim0[15:0]
  g1[2] = (int)((((unsigned)td0 >> 16) & 0xFFFFu)       // tensor_dim0[31:16]
                | (((unsigned)td1 & 0xFFFFu) << 16));   // tensor_dim1[15:0]
  g1[3] = (int)((((unsigned)td1 >> 16) & 0xFFFFu)       // tensor_dim1[31:16]
                | (((unsigned)tile_k & 0xFFFFu) << 16));// tile_dim0
  g1[4] = (int)((unsigned)rows & 0xFFFFu);              // tile_dim1 (tile_dim2=0)
  g1[5] = (int)(unsigned)stride_elems;                  // tensor_dim0_stride[31:0]
  g1[6] = 0;                                            // stride hi / dim1_stride lo
  g1[7] = 0;
  v4i z4; z4[0] = z4[1] = z4[2] = z4[3] = 0;
#if __clang_major__ >= 23
  v8i z8; for (int i = 0; i < 8; ++i) z8[i] = 0;
  __builtin_amdgcn_tensor_load_to_lds(g0, g1, z4, z4, z8, 0);
#else
  __builtin_amdgcn_tensor_load_to_lds(g0, g1, z4, z4, 0);
#endif
}

// low 32 bits of a flat pointer into __shared__ == LDS byte offset (ISA 10.2)
__device__ __forceinline__ unsigned int lds_off(const void* p) {
  return (unsigned int)(uintptr_t)p;
}
#endif  // HAVE_TDM

// ---------------------------------------------------------------------------
// Weight prep: fp32 [K,N] -> f16 [N,K] (per layer, blockIdx.z = layer)
// ---------------------------------------------------------------------------
__global__ __launch_bounds__(256)
void transpose_cvt_kernel(_Float16* __restrict__ dst, const float* __restrict__ src,
                          int K, int N) {
  __shared__ float tile[32][33];
  const size_t lofs = (size_t)blockIdx.z * K * N;
  const float* s = src + lofs;
  _Float16*    d = dst + lofs;
  int kb = blockIdx.y * 32, nb = blockIdx.x * 32;
  int tx = threadIdx.x & 31, ty = threadIdx.x >> 5;     // ty = 0..7
  for (int i = 0; i < 32; i += 8)
    tile[ty + i][tx] = s[(size_t)(kb + ty + i) * N + nb + tx];
  __syncthreads();
  for (int i = 0; i < 32; i += 8)
    d[(size_t)(nb + ty + i) * K + kb + tx] = (_Float16)tile[tx][ty + i];
}

// fp32 -> f16 elementwise (tok_emb is already [N,K] for the tied lm_head)
__global__ __launch_bounds__(256)
void cvt_kernel(_Float16* __restrict__ dst, const float* __restrict__ src, int n) {
  int i = (blockIdx.x * 256 + threadIdx.x) * 8;
  if (i + 8 <= n) {
    float4 a = *(const float4*)(src + i);
    float4 b = *(const float4*)(src + i + 4);
    _Float16* d = dst + i;
    d[0] = (_Float16)a.x; d[1] = (_Float16)a.y; d[2] = (_Float16)a.z; d[3] = (_Float16)a.w;
    d[4] = (_Float16)b.x; d[5] = (_Float16)b.y; d[6] = (_Float16)b.z; d[7] = (_Float16)b.w;
  }
}

// ---------------------------------------------------------------------------
// Embedding
// ---------------------------------------------------------------------------
__global__ __launch_bounds__(256)
void embed_kernel(float* __restrict__ x, const float* __restrict__ tok,
                  const float* __restrict__ pos, const int* __restrict__ idx) {
  int row = blockIdx.x;
  int t   = row & (T_DIM - 1);
  int tk  = idx[row];
  int c   = threadIdx.x * 4;
  float4 a = *(const float4*)(tok + (size_t)tk * E_DIM + c);
  float4 p = *(const float4*)(pos + (size_t)t  * E_DIM + c);
  float4 o; o.x = a.x + p.x; o.y = a.y + p.y; o.z = a.z + p.z; o.w = a.w + p.w;
  *(float4*)(x + (size_t)row * E_DIM + c) = o;
}

// ---------------------------------------------------------------------------
// LayerNorm: fp32 in, f16 out
// ---------------------------------------------------------------------------
__global__ __launch_bounds__(256)
void ln_kernel(_Float16* __restrict__ out, const float* __restrict__ x,
               const float* __restrict__ g, const float* __restrict__ b) {
  __shared__ float red[256];
  int row = blockIdx.x, tid = threadIdx.x;
  float4 v = *(const float4*)(x + (size_t)row * E_DIM + tid * 4);
  red[tid] = v.x + v.y + v.z + v.w;
  __syncthreads();
  for (int off = 128; off > 0; off >>= 1) {
    if (tid < off) red[tid] += red[tid + off];
    __syncthreads();
  }
  float mean = red[0] * (1.f / E_DIM);
  __syncthreads();
  float d0 = v.x - mean, d1 = v.y - mean, d2 = v.z - mean, d3 = v.w - mean;
  red[tid] = d0 * d0 + d1 * d1 + d2 * d2 + d3 * d3;
  __syncthreads();
  for (int off = 128; off > 0; off >>= 1) {
    if (tid < off) red[tid] += red[tid + off];
    __syncthreads();
  }
  float rstd = rsqrtf(red[0] * (1.f / E_DIM) + 1e-5f);
  int c = tid * 4;
  float4 gv = *(const float4*)(g + c);
  float4 bv = *(const float4*)(b + c);
  _Float16* o = out + (size_t)row * E_DIM + c;
  o[0] = (_Float16)(d0 * rstd * gv.x + bv.x);
  o[1] = (_Float16)(d1 * rstd * gv.y + bv.y);
  o[2] = (_Float16)(d2 * rstd * gv.z + bv.z);
  o[3] = (_Float16)(d3 * rstd * gv.w + bv.w);
}

// ---------------------------------------------------------------------------
// WMMA GEMM: C[M,N] = A[M,K](f16,[M,K]) x Bt(f16,[N,K]) (+bias)(+relu)(+resid)
// Block tile 128x128, BK=64, 8 waves, wave tile 32x64 (2x4 accumulators).
// OMODE: 0 = f32 out, 1 = f16 out, 2 = f16 transposed out (C^T[N][M], uint4
//        stores: each lane's C fragment holds 8 consecutive M values).
// ---------------------------------------------------------------------------
__device__ __forceinline__ void mma_stage(const _Float16 (*__restrict__ As)[64],
                                          const _Float16 (*__restrict__ Bs)[64],
                                          v8f acc[2][4], int wm, int wn,
                                          int l16, int grp) {
  for (int ks = 0; ks < 64; ks += 32) {
    Frag16 af[2], bf[4];
    for (int mt = 0; mt < 2; ++mt)
      for (int r = 0; r < 8; ++r)
        af[mt].i[r] = *(const int*)&As[wm + mt * 16 + l16][ks + kpair(r, grp)];
    for (int nt = 0; nt < 4; ++nt)
      for (int r = 0; r < 8; ++r)
        bf[nt].i[r] = *(const int*)&Bs[wn + nt * 16 + l16][ks + kpair(r, grp)];
    for (int mt = 0; mt < 2; ++mt)
      for (int nt = 0; nt < 4; ++nt)
        acc[mt][nt] = wmma_f16(af[mt], bf[nt], acc[mt][nt]);
  }
}

template <bool RELU, bool RESID, int OMODE>
__global__ __launch_bounds__(256)
void gemm16(const _Float16* __restrict__ A, const _Float16* __restrict__ Bt,
            const float* __restrict__ bias, const float* __restrict__ resid,
            float* __restrict__ Cf, _Float16* __restrict__ Ch,
            int M, int N, int K) {
  const int tid  = threadIdx.x;
  const int lane = tid & 31;
  const int w    = tid >> 5;
  const int l16  = lane & 15;
  const int grp  = lane >> 4;
  const int n0   = blockIdx.x * 128;
  const int m0   = blockIdx.y * 128;
  const int wm   = (w & 3) * 32;
  const int wn   = (w >> 2) * 64;

  v8f acc[2][4];
  for (int mt = 0; mt < 2; ++mt)
    for (int nt = 0; nt < 4; ++nt) acc[mt][nt] = vzero8();

  const _Float16* Ab = A  + (size_t)m0 * K;
  const _Float16* Bb = Bt + (size_t)n0 * K;
  const int nk = K >> 6;                     // K/64 stages (>=16 always)

#if HAVE_TDM
  __shared__ _Float16 As[2][128][64];        // 2 x 16 KB
  __shared__ _Float16 Bs[2][128][64];        // 2 x 16 KB   (64 KB total)

  if (tid < 32) {                            // wave 0 drives the TDM
    tdm_load_tile(Ab +  0, lds_off(&As[0][0][0]), 64, 128, K, K, M);
    tdm_load_tile(Bb +  0, lds_off(&Bs[0][0][0]), 64, 128, K, K, N);
    tdm_load_tile(Ab + 64, lds_off(&As[1][0][0]), 64, 128, K, K, M);
    tdm_load_tile(Bb + 64, lds_off(&Bs[1][0][0]), 64, 128, K, K, N);
  }
  for (int i = 0; i < nk; ++i) {
    const int buf = i & 1;
    if (tid < 32) {                          // stage i complete?
      if (i + 1 < nk) __builtin_amdgcn_s_wait_tensorcnt((short)2);
      else            __builtin_amdgcn_s_wait_tensorcnt((short)0);
    }
    __syncthreads();
    mma_stage(As[buf], Bs[buf], acc, wm, wn, l16, grp);
    __syncthreads();                         // everyone done reading buf
    if (tid < 32 && i + 2 < nk) {
      const int kb = (i + 2) * 64;
      tdm_load_tile(Ab + kb, lds_off(&As[buf][0][0]), 64, 128, K, K, M);
      tdm_load_tile(Bb + kb, lds_off(&Bs[buf][0][0]), 64, 128, K, K, N);
    }
  }
#else
  __shared__ _Float16 As[128][64];
  __shared__ _Float16 Bs[128][64];
  const int arow = tid >> 1;
  const int acol = (tid & 1) * 32;
  for (int i = 0; i < nk; ++i) {
    const int kb = i * 64;
    for (int c = 0; c < 4; ++c) {
      *(uint4*)&As[arow][acol + c * 8] =
          *(const uint4*)(Ab + (size_t)arow * K + kb + acol + c * 8);
      *(uint4*)&Bs[arow][acol + c * 8] =
          *(const uint4*)(Bb + (size_t)arow * K + kb + acol + c * 8);
    }
    __syncthreads();
    mma_stage(As, Bs, acc, wm, wn, l16, grp);
    __syncthreads();
  }
#endif

  if (OMODE == 2) {
    // transposed f16 store: C^T[n][m], 8 consecutive m per lane -> one uint4
    for (int mt = 0; mt < 2; ++mt)
      for (int nt = 0; nt < 4; ++nt) {
        int n = n0 + wn + nt * 16 + l16;
        size_t basem = (size_t)n * M + (size_t)(m0 + wm + mt * 16 + 8 * grp);
        _Float16 tmp[8];
        for (int r = 0; r < 8; ++r) tmp[r] = (_Float16)acc[mt][nt][r];
        *(uint4*)(Ch + basem) = *(const uint4*)tmp;
      }
  } else {
    for (int mt = 0; mt < 2; ++mt)
      for (int nt = 0; nt < 4; ++nt)
        for (int r = 0; r < 8; ++r) {
          int m = m0 + wm + mt * 16 + r + 8 * grp;
          int n = n0 + wn + nt * 16 + l16;
          float val = acc[mt][nt][r];
          if (bias) val += bias[n];
          if (RELU) val = fmaxf(val, 0.f);
          if (RESID) val += resid[(size_t)m * N + n];
          if (OMODE == 1) Ch[(size_t)m * N + n] = (_Float16)val;
          else            Cf[(size_t)m * N + n] = val;
        }
  }
}

// ---------------------------------------------------------------------------
// Flash attention: each wave owns 16 query rows of one (b,h); keys streamed 32
// at a time; online softmax; P converted C->A layout through per-wave LDS.
// V is pre-transposed: vT[(hh*64+hs)*2048 + b*1024 + t]  (pairs contiguous in
// t == the WMMA K dimension, so V fragments are 8 b32 loads like Q/K).
// ---------------------------------------------------------------------------
__global__ __launch_bounds__(256)
void attn_kernel(_Float16* __restrict__ o, const _Float16* __restrict__ q,
                 const _Float16* __restrict__ k, const _Float16* __restrict__ vT) {
  __shared__ _Float16 plds[8][16][32];

  const int lane = threadIdx.x & 31;
  const int w    = threadIdx.x >> 5;
  const int tile = blockIdx.x * 8 + w;
  const int qt   = tile & 63;
  const int bh   = tile >> 6;
  const int b    = bh >> 4;
  const int hh   = bh & 15;
  const int l16  = lane & 15;
  const int grp  = lane >> 4;
  const int q0   = qt * 16;
  const size_t base  = ((size_t)b * T_DIM) * E_DIM + hh * HS_DIM;
  const size_t vbase = ((size_t)hh * HS_DIM) * ROWS_DIM + (size_t)b * T_DIM;

  Frag16 qa[2];
  for (int f = 0; f < 2; ++f)
    for (int r = 0; r < 8; ++r) {
      int kk = f * 32 + kpair(r, grp);
      qa[f].i[r] = *(const int*)(q + base + (size_t)(q0 + l16) * E_DIM + kk);
    }

  float mrow[8], lrow[8];
  for (int r = 0; r < 8; ++r) { mrow[r] = -1e30f; lrow[r] = 0.f; }
  v8f oacc[4];
  for (int c = 0; c < 4; ++c) oacc[c] = vzero8();

  const int nsteps = (q0 + 15) / 32 + 1;
  for (int kt = 0; kt < nsteps; ++kt) {
    const int t0 = kt * 32;

    v8f s[2];
    for (int nh = 0; nh < 2; ++nh) {
      Frag16 kf;
      v8f c = vzero8();
      for (int f = 0; f < 2; ++f) {
        for (int r = 0; r < 8; ++r) {
          int kk = f * 32 + kpair(r, grp);
          kf.i[r] = *(const int*)(k + base +
                                  (size_t)(t0 + nh * 16 + l16) * E_DIM + kk);
        }
        c = wmma_f16(qa[f], kf, c);
      }
      s[nh] = c * 0.125f;
      for (int r = 0; r < 8; ++r) {
        int row = q0 + r + 8 * grp;
        int col = t0 + nh * 16 + l16;
        if (col > row) s[nh][r] = -1e30f;
      }
    }

    for (int r = 0; r < 8; ++r) {
      float mx = fmaxf(s[0][r], s[1][r]);
      for (int off = 8; off >= 1; off >>= 1)
        mx = fmaxf(mx, __shfl_xor(mx, off, 32));
      float nm = fmaxf(mrow[r], mx);
      float al = __expf(mrow[r] - nm);
      float p0 = __expf(s[0][r] - nm);
      float p1 = __expf(s[1][r] - nm);
      s[0][r] = p0; s[1][r] = p1;
      float rs = p0 + p1;
      for (int off = 8; off >= 1; off >>= 1)
        rs += __shfl_xor(rs, off, 32);
      lrow[r] = lrow[r] * al + rs;
      mrow[r] = nm;
      for (int c = 0; c < 4; ++c) oacc[c][r] *= al;
    }

    for (int nh = 0; nh < 2; ++nh)
      for (int r = 0; r < 8; ++r)
        plds[w][r + 8 * grp][nh * 16 + l16] = (_Float16)s[nh][r];

    Frag16 pf;
    for (int r = 0; r < 8; ++r)
      pf.i[r] = *(const int*)&plds[w][l16][kpair(r, grp)];

    // V^T fragments: n = hs (lane), K = key (pairs contiguous in memory)
    for (int c = 0; c < 4; ++c) {
      Frag16 vf;
      for (int r = 0; r < 8; ++r)
        vf.i[r] = *(const int*)(vT + vbase +
                                (size_t)(c * 16 + l16) * ROWS_DIM +
                                t0 + kpair(r, grp));
      oacc[c] = wmma_f16(pf, vf, oacc[c]);
    }
  }

  for (int c = 0; c < 4; ++c)
    for (int r = 0; r < 8; ++r) {
      float val = oacc[c][r] / lrow[r];
      o[base + (size_t)(q0 + r + 8 * grp) * E_DIM + c * 16 + l16] =
          (_Float16)val;
    }
}

// ---------------------------------------------------------------------------
// Host side
// ---------------------------------------------------------------------------
extern "C" void kernel_launch(void* const* d_in, const int* in_sizes, int n_in,
                              void* d_out, int out_size, void* d_ws, size_t ws_size,
                              hipStream_t stream) {
  const int*   idx  = (const int*)  d_in[0];
  const float* tok  = (const float*)d_in[1];
  const float* pos  = (const float*)d_in[2];
  const float* Wq   = (const float*)d_in[3];
  const float* Wk   = (const float*)d_in[4];
  const float* Wv   = (const float*)d_in[5];
  const float* Wo   = (const float*)d_in[6];
  const float* bo   = (const float*)d_in[7];
  const float* W1   = (const float*)d_in[8];
  const float* b1   = (const float*)d_in[9];
  const float* W2   = (const float*)d_in[10];
  const float* b2   = (const float*)d_in[11];
  const float* ln1g = (const float*)d_in[12];
  const float* ln1b = (const float*)d_in[13];
  const float* ln2g = (const float*)d_in[14];
  const float* ln2b = (const float*)d_in[15];
  const float* lnfg = (const float*)d_in[16];
  const float* lnfb = (const float*)d_in[17];
  const float* lmb  = (const float*)d_in[18];
  float* out = (float*)d_out;

  const int Rows = ROWS_DIM;                  // B*T = 2048
  const size_t MB = 1u << 20;

  // workspace carve (~300 MB)
  char* ws = (char*)d_ws;
  float*    x    = (float*)    (ws + 0);            //   8 MB residual (fp32)
  _Float16* h    = (_Float16*) (ws +   8 * MB);     //   4 MB LN out (f16)
  _Float16* qb   = (_Float16*) (ws +  12 * MB);
  _Float16* kb   = (_Float16*) (ws +  16 * MB);
  _Float16* vb   = (_Float16*) (ws +  20 * MB);     //   V^T (f16)
  _Float16* ob   = (_Float16*) (ws +  24 * MB);
  _Float16* ffh  = (_Float16*) (ws +  28 * MB);     //  16 MB
  _Float16* WqT  = (_Float16*) (ws +  44 * MB);     //  16 MB each
  _Float16* WkT  = (_Float16*) (ws +  60 * MB);
  _Float16* WvT  = (_Float16*) (ws +  76 * MB);
  _Float16* WoT  = (_Float16*) (ws +  92 * MB);
  _Float16* W1T  = (_Float16*) (ws + 108 * MB);     //  64 MB
  _Float16* W2T  = (_Float16*) (ws + 172 * MB);     //  64 MB
  _Float16* tokT = (_Float16*) (ws + 236 * MB);     //  64 MB

  const size_t EE  = (size_t)E_DIM * E_DIM;
  const size_t EDF = (size_t)E_DIM * DFF_DIM;

  // ---- weight prep (once per launch) ----
  const dim3 gTE(E_DIM / 32, E_DIM / 32, 8);
  transpose_cvt_kernel<<<gTE, 256, 0, stream>>>(WqT, Wq, E_DIM, E_DIM);
  transpose_cvt_kernel<<<gTE, 256, 0, stream>>>(WkT, Wk, E_DIM, E_DIM);
  transpose_cvt_kernel<<<gTE, 256, 0, stream>>>(WvT, Wv, E_DIM, E_DIM);
  transpose_cvt_kernel<<<gTE, 256, 0, stream>>>(WoT, Wo, E_DIM, E_DIM);
  transpose_cvt_kernel<<<dim3(DFF_DIM / 32, E_DIM / 32, 8), 256, 0, stream>>>(
      W1T, W1, E_DIM, DFF_DIM);
  transpose_cvt_kernel<<<dim3(E_DIM / 32, DFF_DIM / 32, 8), 256, 0, stream>>>(
      W2T, W2, DFF_DIM, E_DIM);
  {
    int n = 32000 * E_DIM;
    cvt_kernel<<<n / (256 * 8), 256, 0, stream>>>(tokT, tok, n);
  }

  embed_kernel<<<Rows, 256, 0, stream>>>(x, tok, pos, idx);

  const dim3 gE(E_DIM / 128, Rows / 128);     // (8,16)
  const dim3 gF(DFF_DIM / 128, Rows / 128);   // (32,16)
  const dim3 gV(32000 / 128, Rows / 128);     // (250,16)

  for (int l = 0; l < 8; ++l) {
    ln_kernel<<<Rows, 256, 0, stream>>>(h, x, ln1g + l * E_DIM, ln1b + l * E_DIM);

    gemm16<false, false, 1><<<gE, 256, 0, stream>>>(
        h, WqT + l * EE, nullptr, nullptr, nullptr, qb, Rows, E_DIM, E_DIM);
    gemm16<false, false, 1><<<gE, 256, 0, stream>>>(
        h, WkT + l * EE, nullptr, nullptr, nullptr, kb, Rows, E_DIM, E_DIM);
    gemm16<false, false, 2><<<gE, 256, 0, stream>>>(       // V -> V^T
        h, WvT + l * EE, nullptr, nullptr, nullptr, vb, Rows, E_DIM, E_DIM);

    attn_kernel<<<256, 256, 0, stream>>>(ob, qb, kb, vb);

    gemm16<false, true, 0><<<gE, 256, 0, stream>>>(
        ob, WoT + l * EE, bo + l * E_DIM, x, x, nullptr, Rows, E_DIM, E_DIM);

    ln_kernel<<<Rows, 256, 0, stream>>>(h, x, ln2g + l * E_DIM, ln2b + l * E_DIM);

    gemm16<true, false, 1><<<gF, 256, 0, stream>>>(
        h, W1T + l * EDF, b1 + l * DFF_DIM, nullptr, nullptr, ffh,
        Rows, DFF_DIM, E_DIM);
    gemm16<false, true, 0><<<gE, 256, 0, stream>>>(
        ffh, W2T + l * EDF, b2 + l * E_DIM, x, x, nullptr,
        Rows, E_DIM, DFF_DIM);
  }

  ln_kernel<<<Rows, 256, 0, stream>>>(h, x, lnfg, lnfb);

  // tied lm_head: logits = xf @ tok_emb^T + lm_b (tokT already [N,K] f16)
  gemm16<false, false, 0><<<gV, 256, 0, stream>>>(
      h, tokT, lmb, nullptr, out, nullptr, Rows, 32000, E_DIM);
}
